// NeuralMartingale_36378372997328
// MI455X (gfx1250) — compile-verified, matrix operand order checked by reading, not run on hardware
//
#include <hip/hip_runtime.h>
#include <hip/hip_bf16.h>
#include <stdint.h>

// ---------------------------------------------------------------------------
// Problem constants (from reference)
// ---------------------------------------------------------------------------
#define DXC   16
#define DYC   8
#define MRC   64
#define MBARC 256
#define NTC   9
#define GOUTC 64
#define HXC   1024
#define EXC   512
#define HRNNC 512
#define HBC   2048
#define TOTALC 2049            // (NT-1)*MBAR + 1
#define NROWS (65 * 256)       // base (256) + 64 displaced chunks of 256
#define NTILES (NROWS / 16)    // 1040 row-tiles of 16

typedef __attribute__((ext_vector_type(16))) _Float16 v16h;
typedef __attribute__((ext_vector_type(8)))  _Float16 v8h;
typedef __attribute__((ext_vector_type(8)))  float    v8f;

// Branch-free tanh on the hardware transcendental pipe:
//   tanh(x) = 1 - 2 / (exp(2x) + 1) = 1 - 2 * rcp(exp2(2*log2e*x) + 1)
// v_exp_f32 saturates to +inf / 0 at the extremes, so rcp gives exact +/-1
// tails with no branches.  Error ~1e-6 absolute, far below f16 storage.
__device__ inline float fast_tanh(float x) {
  float e = __builtin_amdgcn_exp2f(x * 2.8853900817779268f);
  return 1.0f - 2.0f * __builtin_amdgcn_rcpf(e + 1.0f);
}

// ---------------------------------------------------------------------------
// Fragment helpers
//
// 16-bit A-matrix 16x32 layout (ISA 7.12.2): lane L holds row M=L&15.
//   lanes 0-15:  halves 0..7 -> K=0..7,   halves 8..15 -> K=16..23
//   lanes 16-31: halves 0..7 -> K=8..15,  halves 8..15 -> K=24..31
// So per lane: two contiguous 8-half (16B) runs at K = hi8 and K = 16+hi8.
// B-matrix (KxN) mirrors this with N = L&15 and same K mapping; we PRE-PACK
// weights so each lane's 16 halves are contiguous (32B) in memory.
// ---------------------------------------------------------------------------
__device__ inline v16h lds_load_A(const _Float16* sh, int strideH, int row,
                                  int k0, int hi8) {
  const _Float16* p = sh + row * strideH + k0 + hi8;
  v8h lo = *reinterpret_cast<const v8h*>(p);
  v8h hb = *reinterpret_cast<const v8h*>(p + 16);
  v16h a;
#pragma unroll
  for (int i = 0; i < 8; ++i) { a[i] = lo[i]; a[8 + i] = hb[i]; }
  return a;
}

__device__ inline v16h load_B(const _Float16* P, int KT, int kt, int nt,
                              int lane) {
  size_t off = ((((size_t)nt * KT + kt) * 32) + lane) * 16;
  return *reinterpret_cast<const v16h*>(P + off);
}

__device__ inline v8f wmma16x16x32(v16h a, v16h b, v8f c) {
  return __builtin_amdgcn_wmma_f32_16x16x32_f16(false, a, false, b,
                                                (short)0, c, false, false);
}

// ---------------------------------------------------------------------------
// Pack a row-major fp32 weight (Kreal x N) into B-fragment f16 layout with K
// zero-padded to Kpad (multiple of 32).  out[((nt*KT+kt)*32+lane)*16+h].
// ---------------------------------------------------------------------------
__global__ void nm_pack_w(const float* __restrict__ W, _Float16* __restrict__ out,
                          int Kreal, int Kpad, int N) {
  size_t idx = (size_t)blockIdx.x * 256 + threadIdx.x;
  size_t total = (size_t)Kpad * N;
  if (idx >= total) return;
  int h    = (int)(idx & 15);
  int lane = (int)((idx >> 4) & 31);
  size_t t = idx >> 9;                 // tile = nt*KT + kt
  int KT = Kpad >> 5;
  int kt = (int)(t % KT);
  int nt = (int)(t / KT);
  int hi8 = (lane >> 4) * 8;
  int k_local = (h < 8) ? (hi8 + h) : (16 + hi8 + (h - 8));
  int K = kt * 32 + k_local;
  int Nc = nt * 16 + (lane & 15);
  float v = (K < Kreal) ? W[(size_t)K * N + Nc] : 0.0f;
  out[idx] = (_Float16)v;
}

// ---------------------------------------------------------------------------
// RNN: eY = scan over Y[q, k+1:, :] with h = tanh(y@Wih + h@Whh + bh)
// One block, 512 threads (one per hidden unit). Tiny (~4 MFLOP).
// ---------------------------------------------------------------------------
__global__ __launch_bounds__(512)
void nm_rnn(const float* __restrict__ Y, const float* __restrict__ Wih,
            const float* __restrict__ Whh, const float* __restrict__ bh,
            const int* pk, const int* pq, float* __restrict__ eY) {
  __shared__ float sh[HRNNC];
  __shared__ float sy[DYC];
  int t = threadIdx.x;
  sh[t] = 0.0f;
  int kk = pk[0], q = pq[0];
  __syncthreads();
  int nsteps = NTC - (kk + 1);
  for (int s = 0; s < nsteps; ++s) {
    if (t < DYC) sy[t] = Y[((size_t)q * NTC + (kk + 1 + s)) * DYC + t];
    __syncthreads();
    float acc = bh[t];
#pragma unroll
    for (int d = 0; d < DYC; ++d) acc += sy[d] * Wih[d * HRNNC + t];
#pragma unroll 4
    for (int j = 0; j < HRNNC; ++j) acc += sh[j] * Whh[j * HRNNC + t];
    float hn = fast_tanh(acc);
    __syncthreads();
    sh[t] = hn;
    __syncthreads();
  }
  eY[t] = sh[t];
}

// ---------------------------------------------------------------------------
// c2048[n] = bb1[n] + sum_j eY[j] * Wb1[513+j, n]   (row-invariant Wb1 part)
// ---------------------------------------------------------------------------
__global__ __launch_bounds__(256)
void nm_cvec(const float* __restrict__ Wb1, const float* __restrict__ bb1,
             const float* __restrict__ eY, float* __restrict__ c2048) {
  int n = blockIdx.x * 256 + threadIdx.x;
  if (n >= HBC) return;
  float acc = bb1[n];
#pragma unroll 4
  for (int j = 0; j < HRNNC; ++j)
    acc += eY[j] * Wb1[(size_t)(1 + EXC + j) * HBC + n];
  c2048[n] = acc;
}

// ---------------------------------------------------------------------------
// Main fused MLP: one block per 16-row tile of the 16640-row batch.
//   L1: h1 = tanh(Xb @ Wx1 + bx1)           16x1024
//   L2: eX = h1 @ Wx2 + bx2                 16x512
//   L3: h2 = tanh(eX @ Wb1x + c + t*wt)     16x2048 (in 256-col chunks)
//   L4: NN = h2 @ Wb2 + bb2                 16x64   (accumulated per chunk)
// 256 threads = 8 wave32s; all WMMA branches are wave-uniform.
// ---------------------------------------------------------------------------
__global__ __launch_bounds__(256)
void nm_mlp(const float* __restrict__ X, const float* __restrict__ stoich,
            const float* __restrict__ times_t, const float* __restrict__ times_tau,
            const _Float16* __restrict__ pWx1, const float* __restrict__ bx1,
            const _Float16* __restrict__ pWx2, const float* __restrict__ bx2,
            const _Float16* __restrict__ pWb1x, const float* __restrict__ c2048,
            const float* __restrict__ Wb1,   // row 0 = time weight
            const _Float16* __restrict__ pWb2, const float* __restrict__ bb2,
            const int* pk, const int* pkp, const int* pqp,
            float* __restrict__ NN) {
  __shared__ _Float16 sA [16 * 32];    //  1 KB  input rows, K padded to 32
  __shared__ _Float16 sH1[16 * 1024];  // 32 KB  layer-1 activations
  __shared__ _Float16 sEX[16 * 512];   // 16 KB  eX
  __shared__ _Float16 sH2[16 * 256];   //  8 KB  h2 column chunk

  const int tid  = threadIdx.x;
  const int lane = tid & 31;
  const int wv   = tid >> 5;           // 0..7
  const int hi8  = (lane >> 4) * 8;
  const int col  = lane & 15;

  const int tile  = blockIdx.x;        // 0..1039
  const int row0  = tile * 16;
  const int chunk = row0 >> 8;         // 0 = base, 1..64 -> j = chunk-1
  const int mbase = row0 & 255;

  const int kk = pk[0], kp = pkp[0], qp = pqp[0];
  const int base = kp * MBARC;

  // ---- stage Xb rows (f16, K zero-padded to 32) ----
  {
    int i = tid >> 4;                  // row 0..15
    int d = tid & 15;                  // dim 0..15
    float v = X[((size_t)qp * TOTALC + (base + mbase + i)) * DXC + d];
    if (chunk > 0) v += stoich[d * MRC + (chunk - 1)];
    sA[i * 32 + d]      = (_Float16)v;
    sA[i * 32 + 16 + d] = (_Float16)0.0f;
  }
  __syncthreads();

  // ---- L1: 64 N-tiles, 8 per wave, single K step (K=32, padded) ----
  {
    v16h a0 = lds_load_A(sA, 32, col, 0, hi8);
#pragma unroll
    for (int it = 0; it < 8; ++it) {
      int nt = wv * 8 + it, n0 = nt * 16;
      v8f c = {};
      c = wmma16x16x32(a0, load_B(pWx1, 1, 0, nt, lane), c);
      float b = bx1[n0 + col];
#pragma unroll
      for (int r = 0; r < 8; ++r)
        sH1[(r + hi8) * 1024 + n0 + col] = (_Float16)fast_tanh(c[r] + b);
    }
  }
  __syncthreads();

  // ---- L2: 32 N-tiles, 4 per wave, K=1024 -> 32 steps ----
  for (int it = 0; it < 4; ++it) {
    int nt = wv * 4 + it, n0 = nt * 16;
    v8f c = {};
#pragma unroll 4
    for (int kt = 0; kt < 32; ++kt)
      c = wmma16x16x32(lds_load_A(sH1, 1024, col, kt * 32, hi8),
                       load_B(pWx2, 32, kt, nt, lane), c);
    float b = bx2[n0 + col];
#pragma unroll
    for (int r = 0; r < 8; ++r)
      sEX[(r + hi8) * 512 + n0 + col] = (_Float16)(c[r] + b);
  }
  __syncthreads();

  // ---- L3+L4, chunked over 8 slabs of 256 h2-columns ----
  float tval[8];
#pragma unroll
  for (int r = 0; r < 8; ++r)
    tval[r] = times_t[kk] + times_tau[mbase + r + hi8];

  v8f acc = {};
  if (wv < 4) {
    int n0 = wv * 16;
#pragma unroll
    for (int r = 0; r < 8; ++r) acc[r] = bb2[n0 + col];
  }

  for (int cch = 0; cch < 8; ++cch) {
    // L3: 16 local N-tiles -> 2 per wave, K=512 -> 16 steps
#pragma unroll
    for (int it = 0; it < 2; ++it) {
      int ntl = wv * 2 + it;
      int ntg = cch * 16 + ntl;
      int n0g = ntg * 16;
      v8f c = {};
#pragma unroll 4
      for (int kt = 0; kt < 16; ++kt)
        c = wmma16x16x32(lds_load_A(sEX, 512, col, kt * 32, hi8),
                         load_B(pWb1x, 16, kt, ntg, lane), c);
      float cb = c2048[n0g + col];
      float wt = Wb1[n0g + col];              // row 0 of Wb1: time weight
#pragma unroll
      for (int r = 0; r < 8; ++r)
        sH2[(r + hi8) * 256 + ntl * 16 + col] =
            (_Float16)fast_tanh(c[r] + cb + tval[r] * wt);
    }
    __syncthreads();
    // L4 partial: waves 0..3 own one 16-col output tile; K chunk = 256
    if (wv < 4) {
#pragma unroll
      for (int ktl = 0; ktl < 8; ++ktl)
        acc = wmma16x16x32(lds_load_A(sH2, 256, col, ktl * 32, hi8),
                           load_B(pWb2, 64, cch * 8 + ktl, wv, lane), acc);
    }
    __syncthreads();
  }

  if (wv < 4) {
    int n0 = wv * 16;
#pragma unroll
    for (int r = 0; r < 8; ++r)
      NN[(size_t)(row0 + r + hi8) * GOUTC + n0 + col] = acc[r];
  }
}

// ---------------------------------------------------------------------------
// Reduction: partial[j][g] = sum_m scale(j,m) * (NN[(1+j)*256+m][g] - NN[m][g])
// scale = (R[base+m+1,j]-R[base+m,j]) * all_d(X[base+m,d]+stoich[d,j] >= 0)
// ---------------------------------------------------------------------------
__global__ __launch_bounds__(256)
void nm_reduce(const float* __restrict__ X, const float* __restrict__ R,
               const float* __restrict__ stoich, const int* pkp, const int* pqp,
               const float* __restrict__ NN, float* __restrict__ partial) {
  const int j = blockIdx.x;            // 0..63
  const int t = threadIdx.x;
  const int kp = pkp[0], qp = pqp[0];
  const int base = kp * MBARC;
  __shared__ float sc[256];
  {
    int m = t;
    const float* xr = X + ((size_t)qp * TOTALC + (base + m)) * DXC;
    bool ok = true;
#pragma unroll
    for (int d = 0; d < DXC; ++d) ok &= (xr[d] + stoich[d * MRC + j]) >= 0.0f;
    float rd = R[((size_t)qp * TOTALC + (base + m + 1)) * MRC + j]
             - R[((size_t)qp * TOTALC + (base + m)) * MRC + j];
    sc[m] = ok ? rd : 0.0f;
  }
  __syncthreads();
  const int g = t & 63, mq = t >> 6;
  float acc = 0.0f;
#pragma unroll 4
  for (int mm = 0; mm < 64; ++mm) {
    int m = mq * 64 + mm;
    float s = sc[m];
    acc += s * (NN[(size_t)((1 + j) * 256 + m) * GOUTC + g]
              - NN[(size_t)m * GOUTC + g]);
  }
  __shared__ float red[256];
  red[t] = acc;
  __syncthreads();
  if (mq == 0)
    partial[j * GOUTC + g] = red[g] + red[64 + g] + red[128 + g] + red[192 + g];
}

__global__ void nm_final(const float* __restrict__ partial, float* __restrict__ out) {
  int g = threadIdx.x;                 // 64 threads
  float a = 0.0f;
#pragma unroll 4
  for (int j = 0; j < MRC; ++j) a += partial[j * GOUTC + g];
  out[g] = a;
}

// ---------------------------------------------------------------------------
// Host launcher
// ---------------------------------------------------------------------------
static inline size_t align256(size_t x) { return (x + 255) & ~(size_t)255; }

extern "C" void kernel_launch(void* const* d_in, const int* in_sizes, int n_in,
                              void* d_out, int out_size, void* d_ws, size_t ws_size,
                              hipStream_t stream) {
  const float* X      = (const float*)d_in[0];
  const float* Y      = (const float*)d_in[1];
  const float* R      = (const float*)d_in[2];
  const float* stoich = (const float*)d_in[3];
  const float* tt     = (const float*)d_in[4];
  const float* ttau   = (const float*)d_in[5];
  const float* Wx1    = (const float*)d_in[6];
  const float* bx1    = (const float*)d_in[7];
  const float* Wx2    = (const float*)d_in[8];
  const float* bx2    = (const float*)d_in[9];
  const float* Wih    = (const float*)d_in[10];
  const float* Whh    = (const float*)d_in[11];
  const float* bh     = (const float*)d_in[12];
  const float* Wb1    = (const float*)d_in[13];
  const float* bb1    = (const float*)d_in[14];
  const float* Wb2    = (const float*)d_in[15];
  const float* bb2    = (const float*)d_in[16];
  const int*   pk     = (const int*)d_in[17];
  const int*   pkp    = (const int*)d_in[18];
  const int*   pq     = (const int*)d_in[19];
  const int*   pqp    = (const int*)d_in[20];
  float* out = (float*)d_out;

  // workspace carve-up
  char* ws = (char*)d_ws;
  size_t off = 0;
  float* eY = (float*)(ws + off);           off = align256(off + HRNNC * 4);
  float* cv = (float*)(ws + off);           off = align256(off + HBC * 4);
  float* partial = (float*)(ws + off);      off = align256(off + MRC * GOUTC * 4);
  float* NN = (float*)(ws + off);           off = align256(off + (size_t)NROWS * GOUTC * 4);
  _Float16* pWx1  = (_Float16*)(ws + off);  off = align256(off + (size_t)32   * HXC * 2);
  _Float16* pWx2  = (_Float16*)(ws + off);  off = align256(off + (size_t)HXC  * EXC * 2);
  _Float16* pWb1x = (_Float16*)(ws + off);  off = align256(off + (size_t)EXC  * HBC * 2);
  _Float16* pWb2  = (_Float16*)(ws + off);  off = align256(off + (size_t)HBC  * GOUTC * 2);
  (void)ws_size; (void)in_sizes; (void)n_in; (void)out_size;

  // 1) pack weights into WMMA B-fragment layout (f16)
  nm_pack_w<<<(32 * HXC) / 256, 256, 0, stream>>>(Wx1, pWx1, DXC, 32, HXC);
  nm_pack_w<<<(HXC * EXC) / 256, 256, 0, stream>>>(Wx2, pWx2, HXC, HXC, EXC);
  nm_pack_w<<<(EXC * HBC) / 256, 256, 0, stream>>>(Wb1 + HBC, pWb1x, EXC, EXC, HBC);
  nm_pack_w<<<(HBC * GOUTC) / 256, 256, 0, stream>>>(Wb2, pWb2, HBC, HBC, GOUTC);

  // 2) RNN -> eY ; 3) row-invariant Wb1 contribution
  nm_rnn<<<1, 512, 0, stream>>>(Y, Wih, Whh, bh, pk, pq, eY);
  nm_cvec<<<HBC / 256, 256, 0, stream>>>(Wb1, bb1, eY, cv);

  // 4) fused WMMA MLP over all 16640 rows
  nm_mlp<<<NTILES, 256, 0, stream>>>(X, stoich, tt, ttau,
                                     pWx1, bx1, pWx2, bx2,
                                     pWb1x, cv, Wb1, pWb2, bb2,
                                     pk, pkp, pqp, NN);

  // 5) deterministic two-stage reduction
  nm_reduce<<<MRC, 256, 0, stream>>>(X, R, stoich, pkp, pqp, NN, partial);
  nm_final<<<1, 64, 0, stream>>>(partial, out);
}
